// MulitHeadAttentionAttLayer_89008902242430
// MI455X (gfx1250) — matrix operand order repaired and checked by reading, not compiled, not run on hardware
//
#include <hip/hip_runtime.h>
#include <cstdint>
#include <cstddef>

// ---------------------------------------------------------------------------
// MI455X (gfx1250) implementation.
// Pipeline (all f32 accumulate, f16 WMMA operands):
//   k_cast_text : text f32 -> f16
//   k_prep_wt   : WQ/Wk/Wv -> transposed f16 [e][d]
//   k_prep_weff : W_effT[l][j] = sum_h W[h*128+j][l]   (folds the tile(1,1,8))
//   k_qkv       : WMMA GEMM (A-tile staged via TDM tensor_load_to_lds)
//   k_scores    : WMMA GEMM S[b] = K^T Q / sqrt(128), k-split x8 partials
//   k_softmax   : reduce partials + softmax over batch axis (axis 0)
//   k_m         : Mt[b][l][i] = sum_j A[b][i][j] * W_effT[l][j]  (tiny, VALU)
//   k_out       : WMMA GEMM out[b] = V[b] @ M[b] (A-tile staged via TDM)
// ---------------------------------------------------------------------------

typedef __attribute__((ext_vector_type(16))) _Float16     v16h;
typedef __attribute__((ext_vector_type(8)))  float        v8f;
typedef __attribute__((ext_vector_type(4)))  unsigned int v4u;
typedef __attribute__((ext_vector_type(8)))  unsigned int v8u;

#define DEVFN static __device__ __forceinline__

namespace {
constexpr int Bc  = 8;
constexpr int Sc  = 4096;
constexpr int Dc  = 128;
constexpr int HDc = 1024;
constexpr int BSc = Bc * Sc;            // 32768 rows total

// workspace layout (bytes)
constexpr size_t OFF_TEXTH = 0;                       // [BS][D] f16
constexpr size_t SZ_TEXTH  = (size_t)BSc * Dc * 2;
constexpr size_t OFF_QT    = OFF_TEXTH + SZ_TEXTH;    // [B][D][S] f16 (transposed)
constexpr size_t SZ_BDS    = (size_t)Bc * Dc * Sc * 2;
constexpr size_t OFF_KT    = OFF_QT + SZ_BDS;         // [B][D][S] f16 (transposed)
constexpr size_t OFF_V     = OFF_KT + SZ_BDS;         // [B][S][D] f16 (row major)
constexpr size_t OFF_WQT   = OFF_V + SZ_BDS;          // [e][d] f16
constexpr size_t SZ_WT     = (size_t)Dc * Dc * 2;
constexpr size_t OFF_WKT   = OFF_WQT + SZ_WT;
constexpr size_t OFF_WVT   = OFF_WKT + SZ_WT;
constexpr size_t OFF_WEFFT = OFF_WVT + SZ_WT;         // [l][j] f32
constexpr size_t SZ_WEFFT  = (size_t)HDc * Dc * 4;
constexpr size_t OFF_SPART = OFF_WEFFT + SZ_WEFFT;    // [8][B][D][D] f32 (k-split partials)
constexpr size_t SZ_SPART  = (size_t)8 * Bc * Dc * Dc * 4;
constexpr size_t OFF_ASOFT = OFF_SPART + SZ_SPART;    // [B][D][D] f32
constexpr size_t SZ_ASOFT  = (size_t)Bc * Dc * Dc * 4;
constexpr size_t OFF_MT    = OFF_ASOFT + SZ_ASOFT;    // [B][HD][D] f16 (transposed M)
} // namespace

DEVFN unsigned short f2h_bits(float f) {
  _Float16 h = (_Float16)f;
  return __builtin_bit_cast(unsigned short, h);
}
DEVFN unsigned int pack2(float a, float b) {
  return (unsigned int)f2h_bits(a) | ((unsigned int)f2h_bits(b) << 16);
}

union FragU { uint4 q[2]; v16h v; };

// Build a 16x(f16) operand fragment from two 16B loads (global or LDS).
DEVFN v16h load_frag16(const unsigned short* p0, const unsigned short* p1) {
  FragU u;
  u.q[0] = *(const uint4*)p0;
  u.q[1] = *(const uint4*)p1;
  return u.v;
}

// ---------------------------------------------------------------------------
// Tensor Data Mover: stage a 2D tile of u16 elements (rows x row_elems,
// row stride == row_elems, i.e. contiguous) from global memory into LDS.
// 2-SGPR-group descriptor form (tensors up to 2D): VADDR2/VADDR3 are NULL.
// Descriptor groups passed as ext_vector values so the backend can allocate
// them directly as SReg_128 / SReg_256 tuples ('s' constraint cannot take
// indirect (struct) operands).
DEVFN void tdm_load_tile_u16(const void* gptr, unsigned lds_off,
                             unsigned rows, unsigned row_elems) {
  const unsigned long long ga = (unsigned long long)(uintptr_t)gptr;
  v4u g0;
  g0[0] = 1u;                                                  // count=1 valid descriptor
  g0[1] = lds_off;                                             // lds_addr (bytes)
  g0[2] = (unsigned)ga;                                        // global_addr[31:0]
  g0[3] = (unsigned)((ga >> 32) & 0x1FFFFFFull) | (2u << 30);  // addr[56:32] | type=2
  v8u g1;
  g1[0] = (1u << 16);          // workgroup_mask=0 | data_size=1 (2 bytes)
  g1[1] = (row_elems << 16);   // tensor_dim0[15:0]   (bits 79:48)
  g1[2] = (rows << 16);        // tensor_dim1[15:0]   (bits 111:80)
  g1[3] = (row_elems << 16);   // tile_dim0           (bits 127:112)
  g1[4] = rows;                // tile_dim1 (bits 143:128), tile_dim2=0
  g1[5] = row_elems;           // tensor_dim0_stride[31:0] (bits 207:160)
  g1[6] = 0u;                  // stride hi / tensor_dim1_stride lo
  g1[7] = 0u;                  // tensor_dim1_stride hi
  asm volatile("tensor_load_to_lds %0, %1" :: "s"(g0), "s"(g1) : "memory");
}

// ---------------------------------------------------------------------------
__global__ void k_cast_text(const float* __restrict__ text,
                            unsigned short* __restrict__ th) {
  int i = blockIdx.x * blockDim.x + threadIdx.x;   // over BS*D/2
  float2 v = ((const float2*)text)[i];
  ((unsigned int*)th)[i] = pack2(v.x, v.y);
}

__global__ void k_prep_wt(const float* __restrict__ WQ,
                          const float* __restrict__ Wk,
                          const float* __restrict__ Wv,
                          unsigned short* __restrict__ wqt,
                          unsigned short* __restrict__ wkt,
                          unsigned short* __restrict__ wvt) {
  int tid = blockIdx.x * blockDim.x + threadIdx.x; // 0..16383
  int d = tid & (Dc - 1);
  int e = tid >> 7;
  wqt[e * Dc + d] = f2h_bits(WQ[d * Dc + e]);
  wkt[e * Dc + d] = f2h_bits(Wk[d * Dc + e]);
  wvt[e * Dc + d] = f2h_bits(Wv[d * Dc + e]);
}

__global__ void k_prep_weff(const float* __restrict__ W,
                            float* __restrict__ wet) {
  int tid = blockIdx.x * blockDim.x + threadIdx.x; // 0..131071
  int j = tid & (Dc - 1);
  int l = tid >> 7;
  float acc = 0.f;
  #pragma unroll
  for (int h = 0; h < 8; ++h) acc += W[(size_t)(h * Dc + j) * HDc + l];
  wet[(size_t)l * Dc + j] = acc;
}

// ---------------------------------------------------------------------------
// QKV projections. Block = 8 waves sharing one 16-row text tile (staged in LDS
// by the TDM); wave w computes the 16x16 Q/K/V tiles for column-tile w.
__global__ void k_qkv(const unsigned short* __restrict__ th,
                      const unsigned short* __restrict__ wqt,
                      const unsigned short* __restrict__ wkt,
                      const unsigned short* __restrict__ wvt,
                      unsigned short* __restrict__ Qt,
                      unsigned short* __restrict__ Kt,
                      unsigned short* __restrict__ Vb) {
  __shared__ __align__(16) unsigned short shA[16 * Dc];   // 4 KB text tile

  const int lane = threadIdx.x & 31;
  const int nt   = threadIdx.x >> 5;   // column tile 0..7
  const int mt   = blockIdx.x;         // row tile 0..2047
  const int n16  = lane & 15;
  const int hf   = lane >> 4;

  if (nt == 0) {
    tdm_load_tile_u16(th + (size_t)mt * 16 * Dc,
                      (unsigned)(uintptr_t)&shA[0], 16, Dc);
    __builtin_amdgcn_s_wait_tensorcnt(0);
  }
  __syncthreads();

  v8f aq = {}; v8f ak = {}; v8f av = {};

  #pragma unroll
  for (int ks = 0; ks < 4; ++ks) {
    const int k0 = ks * 32;
    const unsigned short* ap = shA + n16 * Dc + k0 + hf * 8;
    v16h A = load_frag16(ap, ap + 16);           // K runs {0..7},{16..23} (+half*8)
    const int bcol = nt * 16 + n16;
    const unsigned short* bq = wqt + (size_t)bcol * Dc + k0 + hf * 8;
    const unsigned short* bk = wkt + (size_t)bcol * Dc + k0 + hf * 8;
    const unsigned short* bv = wvt + (size_t)bcol * Dc + k0 + hf * 8;
    v16h Bq = load_frag16(bq, bq + 16);          // K runs {0..7},{16..23} (+half*8)
    v16h Bk = load_frag16(bk, bk + 16);
    v16h Bv = load_frag16(bv, bv + 16);
    aq = __builtin_amdgcn_wmma_f32_16x16x32_f16(false, A, false, Bq, (short)0, aq, false, false);
    ak = __builtin_amdgcn_wmma_f32_16x16x32_f16(false, A, false, Bk, (short)0, ak, false, false);
    av = __builtin_amdgcn_wmma_f32_16x16x32_f16(false, A, false, Bv, (short)0, av, false, false);
  }

  const int b   = (mt * 16) >> 12;        // 4096 rows per batch, tiles never cross
  const int smt = (mt * 16) & (Sc - 1);
  const int sb  = smt + hf * 8;           // 8 consecutive s values per lane (D layout)
  const int col = nt * 16 + n16;

  // Q,K stored transposed [b][feature][s]: contiguous 16B store per lane.
  uint4 pq, pk;
  pq.x = pack2(aq[0], aq[1]); pq.y = pack2(aq[2], aq[3]);
  pq.z = pack2(aq[4], aq[5]); pq.w = pack2(aq[6], aq[7]);
  pk.x = pack2(ak[0], ak[1]); pk.y = pack2(ak[2], ak[3]);
  pk.z = pack2(ak[4], ak[5]); pk.w = pack2(ak[6], ak[7]);
  *(uint4*)(Qt + (size_t)b * Dc * Sc + (size_t)col * Sc + sb) = pq;
  *(uint4*)(Kt + (size_t)b * Dc * Sc + (size_t)col * Sc + sb) = pk;

  // V stored row major [b][s][d] (A-operand of the final GEMM).
  unsigned short* vb = Vb + (size_t)b * Sc * Dc + (size_t)sb * Dc + col;
  #pragma unroll
  for (int r = 0; r < 8; ++r) vb[(size_t)r * Dc] = f2h_bits(av[r]);
}

// ---------------------------------------------------------------------------
// Scores: S[b,i,j] = sum_s Kt[b,i,s] * Qt[b,j,s] / sqrt(128), k-split 8 ways.
__global__ void k_scores(const unsigned short* __restrict__ Qt,
                         const unsigned short* __restrict__ Kt,
                         float* __restrict__ Sp) {
  const int t     = blockIdx.x * 8 + (threadIdx.x >> 5);  // 0..4095
  const int split = t & 7;
  const int jt    = (t >> 3) & 7;
  const int it    = (t >> 6) & 7;
  const int b     = t >> 9;
  const int lane  = threadIdx.x & 31;
  const int n16   = lane & 15;
  const int hf    = lane >> 4;

  const unsigned short* Kb = Kt + (size_t)b * Dc * Sc;
  const unsigned short* Qb = Qt + (size_t)b * Dc * Sc;
  const int i = it * 16 + n16;   // A row (feature i)
  const int j = jt * 16 + n16;   // B col (feature j)

  v8f acc = {};
  #pragma unroll 4
  for (int ks = 0; ks < 16; ++ks) {
    const int s0 = split * 512 + ks * 32;
    const unsigned short* ap = Kb + (size_t)i * Sc + s0 + hf * 8;
    v16h A = load_frag16(ap, ap + 16);
    const unsigned short* bp = Qb + (size_t)j * Sc + s0 + hf * 8;
    v16h Bf = load_frag16(bp, bp + 16);
    acc = __builtin_amdgcn_wmma_f32_16x16x32_f16(false, A, false, Bf, (short)0, acc, false, false);
  }

  const float scale = 0.08838834764831845f;   // 1/sqrt(128)
  float* out = Sp + (size_t)split * Bc * Dc * Dc + (size_t)b * Dc * Dc;
  #pragma unroll
  for (int r = 0; r < 8; ++r) {
    const int ii = it * 16 + r + 8 * hf;
    out[(size_t)ii * Dc + jt * 16 + n16] = acc[r] * scale;
  }
}

// Softmax over the batch axis (axis 0), reducing the 8 k-split partials first.
__global__ void k_softmax(const float* __restrict__ Sp,
                          float* __restrict__ A) {
  const int p = blockIdx.x * blockDim.x + threadIdx.x;  // 0..16383 over (i,j)
  float v[Bc];
  float mx = -3.4e38f;
  #pragma unroll
  for (int b = 0; b < Bc; ++b) {
    float s = 0.f;
    #pragma unroll
    for (int sp = 0; sp < 8; ++sp)
      s += Sp[(size_t)sp * Bc * Dc * Dc + (size_t)b * Dc * Dc + p];
    v[b] = s;
    mx = fmaxf(mx, s);
  }
  float sum = 0.f;
  #pragma unroll
  for (int b = 0; b < Bc; ++b) { v[b] = __expf(v[b] - mx); sum += v[b]; }
  const float inv = 1.f / sum;
  #pragma unroll
  for (int b = 0; b < Bc; ++b) A[(size_t)b * Dc * Dc + p] = v[b] * inv;
}

// Mt[b][l][i] = sum_j A[b][i][j] * W_effT[l][j]  (0.27 GFLOP; plain fp32 VALU)
__global__ void k_m(const float* __restrict__ A,
                    const float* __restrict__ wet,
                    unsigned short* __restrict__ Mt) {
  const int tid = blockIdx.x * blockDim.x + threadIdx.x;  // 0..(8*1024*128)
  const int i = tid & (Dc - 1);
  const int l = (tid >> 7) & (HDc - 1);
  const int b = tid >> 17;
  const float4* ar = (const float4*)(A + (size_t)b * Dc * Dc + (size_t)i * Dc);
  const float4* wr = (const float4*)(wet + (size_t)l * Dc);
  float acc = 0.f;
  #pragma unroll 8
  for (int j = 0; j < Dc / 4; ++j) {
    const float4 a = ar[j], w = wr[j];
    acc += a.x * w.x + a.y * w.y + a.z * w.z + a.w * w.w;
  }
  Mt[(size_t)b * HDc * Dc + (size_t)l * Dc + i] = f2h_bits(acc);
}

// ---------------------------------------------------------------------------
// Final: out[b][s][l] = sum_i V[b][s][i] * Mt[b][l][i]   (dominant GEMM).
// Block = 8 waves sharing one 16-row V tile staged in LDS by the TDM.
__global__ void k_out(const unsigned short* __restrict__ Vb,
                      const unsigned short* __restrict__ Mt,
                      float* __restrict__ out) {
  __shared__ __align__(16) unsigned short shA[16 * Dc];   // 4 KB V tile

  const int mt   = blockIdx.x >> 3;                       // row tile 0..2047
  const int nt   = ((blockIdx.x & 7) << 3) + (threadIdx.x >> 5);  // 0..63
  const int lane = threadIdx.x & 31;
  const int n16  = lane & 15;
  const int hf   = lane >> 4;

  const int b   = (mt * 16) >> 12;
  const int smt = (mt * 16) & (Sc - 1);
  const unsigned short* Mb = Mt + (size_t)b * HDc * Dc;

  if ((threadIdx.x >> 5) == 0) {
    tdm_load_tile_u16(Vb + (size_t)b * Sc * Dc + (size_t)smt * Dc,
                      (unsigned)(uintptr_t)&shA[0], 16, Dc);
    __builtin_amdgcn_s_wait_tensorcnt(0);
  }
  __syncthreads();

  const int l = nt * 16 + n16;

  v8f acc = {};
  #pragma unroll
  for (int ks = 0; ks < 4; ++ks) {
    const int k0 = ks * 32;
    const unsigned short* ap = shA + n16 * Dc + k0 + hf * 8;
    v16h A = load_frag16(ap, ap + 16);
    const unsigned short* bp = Mb + (size_t)l * Dc + k0 + hf * 8;
    v16h Bf = load_frag16(bp, bp + 16);
    acc = __builtin_amdgcn_wmma_f32_16x16x32_f16(false, A, false, Bf, (short)0, acc, false, false);
  }

  #pragma unroll
  for (int r = 0; r < 8; ++r) {
    const int row = mt * 16 + r + 8 * hf;       // global row in [0, B*S)
    out[(size_t)row * HDc + l] = acc[r];
  }
}

// ---------------------------------------------------------------------------
extern "C" void kernel_launch(void* const* d_in, const int* in_sizes, int n_in,
                              void* d_out, int out_size, void* d_ws, size_t ws_size,
                              hipStream_t stream) {
  (void)in_sizes; (void)n_in; (void)out_size; (void)ws_size;

  const float* text = (const float*)d_in[0];
  const float* WQ   = (const float*)d_in[1];
  const float* Wk   = (const float*)d_in[2];
  const float* Wv   = (const float*)d_in[3];
  const float* W    = (const float*)d_in[4];
  float* out = (float*)d_out;

  char* ws = (char*)d_ws;
  unsigned short* th    = (unsigned short*)(ws + OFF_TEXTH);
  unsigned short* Qt    = (unsigned short*)(ws + OFF_QT);
  unsigned short* Kt    = (unsigned short*)(ws + OFF_KT);
  unsigned short* Vb    = (unsigned short*)(ws + OFF_V);
  unsigned short* wqt   = (unsigned short*)(ws + OFF_WQT);
  unsigned short* wkt   = (unsigned short*)(ws + OFF_WKT);
  unsigned short* wvt   = (unsigned short*)(ws + OFF_WVT);
  float*          wet   = (float*)(ws + OFF_WEFFT);
  float*          Sp    = (float*)(ws + OFF_SPART);
  float*          Asoft = (float*)(ws + OFF_ASOFT);
  unsigned short* Mt    = (unsigned short*)(ws + OFF_MT);

  k_cast_text<<<(BSc * Dc / 2) / 256, 256, 0, stream>>>(text, th);
  k_prep_wt  <<<(Dc * Dc) / 256, 256, 0, stream>>>(WQ, Wk, Wv, wqt, wkt, wvt);
  k_prep_weff<<<(HDc * Dc) / 256, 256, 0, stream>>>(W, wet);
  k_qkv      <<<BSc / 16, 256, 0, stream>>>(th, wqt, wkt, wvt, Qt, Kt, Vb);
  k_scores   <<<512, 256, 0, stream>>>(Qt, Kt, Sp);
  k_softmax  <<<(Dc * Dc) / 256, 256, 0, stream>>>(Sp, Asoft);
  k_m        <<<(Bc * HDc * Dc) / 256, 256, 0, stream>>>(Asoft, wet, Mt);
  k_out      <<<(BSc / 16) * (HDc / 16) / 8, 256, 0, stream>>>(Vb, Mt, out);
}
